// InterferenceBypassSelfAttention_37203006718038
// MI455X (gfx1250) — matrix operand order repaired and checked by reading, not compiled, not run on hardware
//
#include <hip/hip_runtime.h>
#include <cstdint>

typedef __attribute__((ext_vector_type(16))) _Float16 v16h;
typedef __attribute__((ext_vector_type(8)))  float    v8f;

union FragH { v16h v; unsigned int u[8]; };
union PackH { _Float16 h[2]; unsigned int u; };

#define D_MODEL 1024
#define N_HEAD  16
#define HEAD_DIM 64
#define SEQ 2048
#define BSZ 2
#define MROWS (BSZ*SEQ)           // 4096

// A-matrix 16x32 f16 per-lane K pattern (ISA 7.12.2, dense A table)
__device__ __forceinline__ int kpatA(int v, int half) {
  return 2*(v&3) + 8*half + 16*(v>>2);
}
// B-matrix 32x16 f16 per-lane K pattern (same as A: lanes0-15: K=0..7,16..23; lanes16-31: K=8..15,24..31)
__device__ __forceinline__ int kpatB(int v, int half) {
  return 2*(v&3) + 8*half + 16*(v>>2);
}

__device__ __forceinline__ v8f wmma_f16(const v16h& a, const v16h& b, const v8f& c) {
  return __builtin_amdgcn_wmma_f32_16x16x32_f16(false, a, false, b, (short)0, c, false, false);
}

// ---- CDNA5 async global->LDS copy (ASYNCcnt path), inline asm per bridge doc ----
__device__ __forceinline__ unsigned lds_off_u32(const void* p) {
  // low 32 bits of an LDS generic address are the LDS byte offset (ISA 10.2 aperture rules)
  return (unsigned)(uintptr_t)p;
}
__device__ __forceinline__ void async_copy_b128(unsigned lds_byte_off, const void* gaddr) {
  asm volatile("global_load_async_to_lds_b128 %0, %1, off"
               :: "v"(lds_byte_off), "v"(gaddr) : "memory");
}
__device__ __forceinline__ void wait_async_le4() {
  asm volatile("s_wait_asynccnt 0x4" ::: "memory");
}
__device__ __forceinline__ void wait_async_0() {
  asm volatile("s_wait_asynccnt 0x0" ::: "memory");
}

// ---------------- f32 -> f16 convert ----------------
__global__ void cvt_f16_kernel(const float* __restrict__ in, _Float16* __restrict__ out, int n) {
  int i = blockIdx.x * blockDim.x + threadIdx.x;
  int stride = gridDim.x * blockDim.x;
  for (; i < n; i += stride) out[i] = (_Float16)in[i];
}

// ---------------- WMMA GEMM, async-LDS double buffered ----------------
// C[M,N] = A[M,K] * W[N,K]^T
// MODE 0: f32 row-major [M,N].  MODE 1: f16 transposed head-split V^T [BH][hd][SEQ].
template<int MODE>
__global__ __launch_bounds__(256) void gemm_f16_kernel(const _Float16* __restrict__ A,
                                                       const _Float16* __restrict__ Bw,
                                                       void* __restrict__ Cout,
                                                       int M, int N, int K) {
  __shared__ __align__(16) _Float16 ldsA[2][128 * 32];   // 2 x 8 KB
  __shared__ __align__(16) _Float16 ldsB[2][128 * 32];   // 2 x 8 KB

  const int tid  = threadIdx.x;
  const int lane = tid & 31;
  const int wave = tid >> 5;
  const int lm = lane & 15, lh = lane >> 4;
  const int wm = wave >> 2, wn = wave & 3;              // 2x4 wave grid
  const int rowBase = blockIdx.y * 128;
  const int colBase = blockIdx.x * 128;

  // async-copy addressing: 256 threads x 16B = 4KB = 64 rows of 32 f16 per issue
  const int cr = tid >> 2;     // 0..63 (row within group)
  const int cc = tid & 3;      // 16B chunk within 64B row

  auto issue_tiles = [&](int buf, int kk) {
#pragma unroll
    for (int g = 0; g < 2; ++g) {
      int r = g * 64 + cr;
      const _Float16* ga = A  + (size_t)(rowBase + r) * K + kk + cc * 8;
      async_copy_b128(lds_off_u32(&ldsA[buf][r * 32 + cc * 8]), ga);
    }
#pragma unroll
    for (int g = 0; g < 2; ++g) {
      int r = g * 64 + cr;
      const _Float16* gb = Bw + (size_t)(colBase + r) * K + kk + cc * 8;
      async_copy_b128(lds_off_u32(&ldsB[buf][r * 32 + cc * 8]), gb);
    }
  };

  v8f acc[4][2] = {};
  const int nk = K / 32;
  issue_tiles(0, 0);
  for (int t = 0; t < nk; ++t) {
    const int buf = t & 1;
    if (t + 1 < nk) { issue_tiles(buf ^ 1, (t + 1) * 32); wait_async_le4(); }
    else            { wait_async_0(); }
    __syncthreads();

    const _Float16* la = &ldsA[buf][0];
    const _Float16* lb = &ldsB[buf][0];
    FragH af[4], bf[2];
#pragma unroll
    for (int mt = 0; mt < 4; ++mt) {
      const _Float16* ap = la + (wm * 64 + mt * 16 + lm) * 32;
#pragma unroll
      for (int v = 0; v < 8; ++v)
        af[mt].u[v] = *reinterpret_cast<const unsigned int*>(ap + kpatA(v, lh));
    }
#pragma unroll
    for (int nt = 0; nt < 2; ++nt) {
      const _Float16* bp = lb + (wn * 32 + nt * 16 + lm) * 32;
#pragma unroll
      for (int v = 0; v < 8; ++v)
        bf[nt].u[v] = *reinterpret_cast<const unsigned int*>(bp + kpatB(v, lh));
    }
#pragma unroll
    for (int mt = 0; mt < 4; ++mt)
#pragma unroll
      for (int nt = 0; nt < 2; ++nt)
        acc[mt][nt] = wmma_f16(af[mt].v, bf[nt].v, acc[mt][nt]);
    __syncthreads();
  }

  // epilogue: C layout: within a 16x16 tile, col = lm, row = r + 8*lh
#pragma unroll
  for (int mt = 0; mt < 4; ++mt)
#pragma unroll
    for (int nt = 0; nt < 2; ++nt)
#pragma unroll
      for (int r = 0; r < 8; ++r) {
        int row = rowBase + wm * 64 + mt * 16 + r + 8 * lh;
        int col = colBase + wn * 32 + nt * 16 + lm;
        float val = acc[mt][nt][r];
        if (MODE == 0) {
          ((float*)Cout)[(size_t)row * N + col] = val;
        } else {
          int b = row >> 11, s = row & 2047;
          int h = col >> 6,  d = col & 63;
          // V^T layout: [(b*16+h)][d][s]
          ((_Float16*)Cout)[(((size_t)(b * N_HEAD + h) * HEAD_DIM + d) << 11) + s] = (_Float16)val;
        }
      }
}

// ---------------- fuse: softplus+RMSnorm amp, pi*tanh phase, emit [cos||sin] f16 ----------------
__device__ __forceinline__ float softplusf(float x) {
  return (x > 20.f) ? x : log1pf(__expf(x));
}

__global__ __launch_bounds__(256) void fuse_qk_kernel(const float* __restrict__ ampraw,
                                                      const float* __restrict__ phiraw,
                                                      _Float16* __restrict__ out2) {
  const int wave = threadIdx.x >> 5, lane = threadIdx.x & 31;
  const int idx = blockIdx.x * 8 + wave;      // over MROWS * N_HEAD
  const int i = idx >> 4;                     // token row (b*SEQ + s)
  const int h = idx & 15;
  const int d0 = lane * 2;
  const float* ap = ampraw + (size_t)i * D_MODEL + h * HEAD_DIM;
  const float* pp = phiraw + (size_t)i * D_MODEL + h * HEAD_DIM;

  float a0 = softplusf(ap[d0]);
  float a1 = softplusf(ap[d0 + 1]);
  float ss = a0*a0 + a1*a1;
#pragma unroll
  for (int m = 1; m < 32; m <<= 1) ss += __shfl_xor(ss, m, 32);
  float rn = rsqrtf(ss * (1.0f / 64.0f) + 1e-6f);
  a0 *= rn; a1 *= rn;

  const float PI = 3.14159265358979323846f;
  float p0 = PI * tanhf(pp[d0]);
  float p1 = PI * tanhf(pp[d0 + 1]);
  float s0, c0, s1, c1;
  __sincosf(p0, &s0, &c0);
  __sincosf(p1, &s1, &c1);

  int b = i >> 11, s = i & 2047;
  _Float16* orow = out2 + ((size_t)(b * N_HEAD + h) * SEQ + s) * 128;
  PackH pc, ps;
  pc.h[0] = (_Float16)(a0 * c0); pc.h[1] = (_Float16)(a1 * c1);
  ps.h[0] = (_Float16)(a0 * s0); ps.h[1] = (_Float16)(a1 * s1);
  *reinterpret_cast<unsigned int*>(orow + d0)      = pc.u;
  *reinterpret_cast<unsigned int*>(orow + 64 + d0) = ps.u;
}

// ---------------- flash attention: Q2[BH,S,128] x K2 -> causal softmax -> x V^T[BH,64,S] ----------------
__global__ __launch_bounds__(128) void flash_attn_kernel(const _Float16* __restrict__ Q2,
                                                         const _Float16* __restrict__ K2,
                                                         const _Float16* __restrict__ Vt,
                                                         const float* __restrict__ lscale,
                                                         _Float16* __restrict__ attn) {
  __shared__ __align__(16) _Float16 pbuf[4][16][64];

  const int bh = blockIdx.y;
  const int b = bh >> 4, h = bh & 15;
  const int qb = blockIdx.x;
  const int wave = threadIdx.x >> 5, lane = threadIdx.x & 31;
  const int lm = lane & 15, lh = lane >> 4;
  const int q0 = qb * 64 + wave * 16;
  const float NEG = -3.0e38f;
  const float scale = __expf(lscale[h]) * 0.125f;   // exp(log_scale)/sqrt(64)

  const _Float16* Qb = Q2 + (size_t)bh * SEQ * 128;
  const _Float16* Kb = K2 + (size_t)bh * SEQ * 128;
  const _Float16* Vb = Vt + (size_t)bh * HEAD_DIM * SEQ;

  // preload Q fragments (16 rows x 128 dims = 4 k-steps)
  FragH qf[4];
#pragma unroll
  for (int ks = 0; ks < 4; ++ks) {
    const _Float16* qp = Qb + (size_t)(q0 + lm) * 128 + ks * 32;
#pragma unroll
    for (int v = 0; v < 8; ++v)
      qf[ks].u[v] = *reinterpret_cast<const unsigned int*>(qp + kpatA(v, lh));
  }

  v8f oacc[4] = {};
  float m_i[8], l_i[8];
#pragma unroll
  for (int r = 0; r < 8; ++r) { m_i[r] = NEG; l_i[r] = 0.f; }

  for (int kb = 0; kb <= qb; ++kb) {
    // S = Q2 * K2^T   (16 x 64 per wave)
    v8f sacc[4] = {};
#pragma unroll
    for (int nt = 0; nt < 4; ++nt) {
      const int key = kb * 64 + nt * 16 + lm;
      const _Float16* kp = Kb + (size_t)key * 128;
#pragma unroll
      for (int ks = 0; ks < 4; ++ks) {
        FragH bf;
#pragma unroll
        for (int v = 0; v < 8; ++v)
          bf.u[v] = *reinterpret_cast<const unsigned int*>(kp + ks * 32 + kpatB(v, lh));
        sacc[nt] = wmma_f16(qf[ks].v, bf.v, sacc[nt]);
      }
    }
    // scale + causal mask + row max
    float rmax[8];
#pragma unroll
    for (int r = 0; r < 8; ++r) rmax[r] = NEG;
    const bool diag = (kb == qb);
#pragma unroll
    for (int nt = 0; nt < 4; ++nt)
#pragma unroll
      for (int r = 0; r < 8; ++r) {
        float s = sacc[nt][r] * scale;
        int key = kb * 64 + nt * 16 + lm;
        int qrow = q0 + r + 8 * lh;
        if (diag && key > qrow) s = NEG;
        sacc[nt][r] = s;
        rmax[r] = fmaxf(rmax[r], s);
      }
#pragma unroll
    for (int m = 1; m < 16; m <<= 1)
#pragma unroll
      for (int r = 0; r < 8; ++r) rmax[r] = fmaxf(rmax[r], __shfl_xor(rmax[r], m, 32));

    float alpha[8], rsum[8];
#pragma unroll
    for (int r = 0; r < 8; ++r) {
      float mn = fmaxf(m_i[r], rmax[r]);
      alpha[r] = __expf(m_i[r] - mn);
      m_i[r] = mn;
      rsum[r] = 0.f;
    }
    // P = exp(S-m), stage through LDS into A-fragment layout
#pragma unroll
    for (int nt = 0; nt < 4; ++nt)
#pragma unroll
      for (int r = 0; r < 8; ++r) {
        float p = __expf(sacc[nt][r] - m_i[r]);
        pbuf[wave][r + 8 * lh][nt * 16 + lm] = (_Float16)p;
        rsum[r] += p;
      }
#pragma unroll
    for (int m = 1; m < 16; m <<= 1)
#pragma unroll
      for (int r = 0; r < 8; ++r) rsum[r] += __shfl_xor(rsum[r], m, 32);
#pragma unroll
    for (int r = 0; r < 8; ++r) l_i[r] = l_i[r] * alpha[r] + rsum[r];
#pragma unroll
    for (int nt = 0; nt < 4; ++nt)
#pragma unroll
      for (int r = 0; r < 8; ++r) oacc[nt][r] *= alpha[r];

    __syncthreads();
    // O += P * V   (K = 64 keys -> 2 k-steps); V^T rows are d, contiguous in key
#pragma unroll
    for (int ks = 0; ks < 2; ++ks) {
      FragH pa;
#pragma unroll
      for (int v = 0; v < 8; ++v)
        pa.u[v] = *reinterpret_cast<const unsigned int*>(&pbuf[wave][lm][ks * 32 + kpatA(v, lh)]);
#pragma unroll
      for (int nt = 0; nt < 4; ++nt) {
        FragH bv;
        const _Float16* vp = Vb + (size_t)(nt * 16 + lm) * SEQ + kb * 64 + ks * 32;
#pragma unroll
        for (int v = 0; v < 8; ++v)
          bv.u[v] = *reinterpret_cast<const unsigned int*>(vp + kpatB(v, lh));
        oacc[nt] = wmma_f16(pa.v, bv.v, oacc[nt]);
      }
    }
    __syncthreads();
  }

  // write out: attn[b, q, h*64 + d] f16 (merged heads)
#pragma unroll
  for (int nt = 0; nt < 4; ++nt)
#pragma unroll
    for (int r = 0; r < 8; ++r) {
      int qrow = q0 + r + 8 * lh;
      float o = oacc[nt][r] / l_i[r];
      attn[((size_t)b * SEQ + qrow) * D_MODEL + h * HEAD_DIM + nt * 16 + lm] = (_Float16)o;
    }
}

// ---------------- host-side orchestration ----------------
extern "C" void kernel_launch(void* const* d_in, const int* in_sizes, int n_in,
                              void* d_out, int out_size, void* d_ws, size_t ws_size,
                              hipStream_t stream) {
  (void)in_sizes; (void)n_in; (void)out_size; (void)ws_size;
  const float* x      = (const float*)d_in[0];
  const float* Wq_amp = (const float*)d_in[1];
  const float* Wk_amp = (const float*)d_in[2];
  const float* Wq_phi = (const float*)d_in[3];
  const float* Wk_phi = (const float*)d_in[4];
  const float* Wv     = (const float*)d_in[5];
  const float* Wo     = (const float*)d_in[6];
  const float* lscale = (const float*)d_in[7];

  char* ws = (char*)d_ws;
  size_t off = 0;
  _Float16* xh   = (_Float16*)(ws + off); off += (size_t)MROWS * D_MODEL * 2;     // 8 MB
  _Float16* wqa  = (_Float16*)(ws + off); off += (size_t)D_MODEL * D_MODEL * 2;
  _Float16* wka  = (_Float16*)(ws + off); off += (size_t)D_MODEL * D_MODEL * 2;
  _Float16* wqp  = (_Float16*)(ws + off); off += (size_t)D_MODEL * D_MODEL * 2;
  _Float16* wkp  = (_Float16*)(ws + off); off += (size_t)D_MODEL * D_MODEL * 2;
  _Float16* wvh  = (_Float16*)(ws + off); off += (size_t)D_MODEL * D_MODEL * 2;
  _Float16* woh  = (_Float16*)(ws + off); off += (size_t)D_MODEL * D_MODEL * 2;   // +12 MB
  float*    raw0 = (float*)(ws + off);    off += (size_t)MROWS * D_MODEL * 4;     // 16 MB
  float*    raw1 = (float*)(ws + off);    off += (size_t)MROWS * D_MODEL * 4;     // 16 MB
  _Float16* Q2   = (_Float16*)(ws + off); off += (size_t)BSZ * N_HEAD * SEQ * 128 * 2; // 16 MB
  _Float16* K2   = (_Float16*)(ws + off); off += (size_t)BSZ * N_HEAD * SEQ * 128 * 2; // 16 MB
  _Float16* Vtd  = (_Float16*)(ws + off); off += (size_t)BSZ * N_HEAD * HEAD_DIM * SEQ * 2; // 8 MB
  _Float16* attn = (_Float16*)(ws + off); off += (size_t)MROWS * D_MODEL * 2;     // 8 MB

  const int NW = D_MODEL * D_MODEL;
  cvt_f16_kernel<<<2048, 256, 0, stream>>>(x, xh, MROWS * D_MODEL);
  cvt_f16_kernel<<<1024, 256, 0, stream>>>(Wq_amp, wqa, NW);
  cvt_f16_kernel<<<1024, 256, 0, stream>>>(Wk_amp, wka, NW);
  cvt_f16_kernel<<<1024, 256, 0, stream>>>(Wq_phi, wqp, NW);
  cvt_f16_kernel<<<1024, 256, 0, stream>>>(Wk_phi, wkp, NW);
  cvt_f16_kernel<<<1024, 256, 0, stream>>>(Wv, wvh, NW);
  cvt_f16_kernel<<<1024, 256, 0, stream>>>(Wo, woh, NW);

  dim3 ggrid(D_MODEL / 128, MROWS / 128);   // (8, 32)
  // Q projections + fuse
  gemm_f16_kernel<0><<<ggrid, 256, 0, stream>>>(xh, wqa, raw0, MROWS, D_MODEL, D_MODEL);
  gemm_f16_kernel<0><<<ggrid, 256, 0, stream>>>(xh, wqp, raw1, MROWS, D_MODEL, D_MODEL);
  fuse_qk_kernel<<<MROWS * N_HEAD / 8, 256, 0, stream>>>(raw0, raw1, Q2);
  // K projections + fuse
  gemm_f16_kernel<0><<<ggrid, 256, 0, stream>>>(xh, wka, raw0, MROWS, D_MODEL, D_MODEL);
  gemm_f16_kernel<0><<<ggrid, 256, 0, stream>>>(xh, wkp, raw1, MROWS, D_MODEL, D_MODEL);
  fuse_qk_kernel<<<MROWS * N_HEAD / 8, 256, 0, stream>>>(raw0, raw1, K2);
  // V projection -> V^T f16 [BH][hd][SEQ]
  gemm_f16_kernel<1><<<ggrid, 256, 0, stream>>>(xh, wvh, Vtd, MROWS, D_MODEL, D_MODEL);
  // flash attention
  flash_attn_kernel<<<dim3(SEQ / 64, BSZ * N_HEAD), 128, 0, stream>>>(Q2, K2, Vtd, lscale, attn);
  // output projection -> f32 d_out [B*S, D]
  gemm_f16_kernel<0><<<ggrid, 256, 0, stream>>>(attn, woh, (float*)d_out, MROWS, D_MODEL, D_MODEL);
}
